// MultiHeadAttention_54623394070801
// MI455X (gfx1250) — compile-verified, matrix operand order checked by reading, not run on hardware
//
#include <hip/hip_runtime.h>

// ---------------------------------------------------------------------------
// Multi-head attention forward for MI455X (gfx1250, wave32, WMMA).
//   E=768, H=12, Dh=64, B=2, T=2048.
// Matrix math: v_wmma_f32_16x16x32_f16.
// Shared tiles staged in LDS via global_load_async_to_lds_b128 (ASYNCcnt),
// double-buffered; pipeline loops are peeled so the steady state is
// branch-free: fill(next) -> s_wait_asynccnt 1 -> barrier -> WMMA burst.
// ---------------------------------------------------------------------------

#define E 768
#define NH 12
#define DH 64
#define BATCH 2
#define SEQ 2048
#define M_TOTAL (BATCH * SEQ)   // 4096 rows for the projection GEMMs
#define KSTEPS (E / 32)         // 24 K-steps for the projection GEMMs

typedef __attribute__((ext_vector_type(16))) _Float16 v16h;
typedef __attribute__((ext_vector_type(8)))  _Float16 v8h;
typedef __attribute__((ext_vector_type(8)))  float    v8f;

// ---------------------------------------------------------------------------
// CDNA5 async copy: 16 bytes global -> LDS, tracked by ASYNCcnt.
// VDST carries the LDS byte offset (low 32 bits of the flat shared address,
// per the LDS aperture rule); VADDR is the 64-bit global address.
// ---------------------------------------------------------------------------
__device__ __forceinline__ void async_copy16(const void* gptr, void* lptr) {
  unsigned lds_off = (unsigned)(unsigned long long)lptr;
  unsigned long long ga = (unsigned long long)gptr;
  asm volatile("global_load_async_to_lds_b128 %0, %1, off"
               :: "v"(lds_off), "v"(ga) : "memory");
}

template <int N>
__device__ __forceinline__ void wait_asynccnt() {
  asm volatile("s_wait_asynccnt %0" :: "i"(N) : "memory");
}

__device__ __forceinline__ void wait_dscnt0() {
  asm volatile("s_wait_dscnt 0" ::: "memory");
}

// ---------------------------------------------------------------------------
// Fragment loaders.  16-bit A-matrix 16x32 layout (ISA 7.12.2): lanes 0-15
// hold M=lane, K={0..7,16..23}; lanes 16-31 hold M=lane-16, K={8..15,24..31}.
// B (32x16) mirrors this with N in place of M when the source is N-major with
// K contiguous.  Works on global or LDS-backed pointers (addrspace inferred).
// ---------------------------------------------------------------------------
__device__ __forceinline__ v16h frag_ld_f16(const _Float16* base, int ld) {
  const int lane = threadIdx.x & 31;
  const int m  = lane & 15;
  const int k0 = (lane & 16) ? 8 : 0;
  const _Float16* p = base + (size_t)m * ld + k0;
  v8h lo = *(const v8h*)(p);
  v8h hi = *(const v8h*)(p + 16);
  v16h r;
#pragma unroll
  for (int i = 0; i < 8; ++i) { r[i] = lo[i]; r[i + 8] = hi[i]; }
  return r;
}

__device__ __forceinline__ v16h frag_ld_f32(const float* base, int ld) {
  const int lane = threadIdx.x & 31;
  const int m  = lane & 15;
  const int k0 = (lane & 16) ? 8 : 0;
  const float* p = base + (size_t)m * ld + k0;
  v16h r;
#pragma unroll
  for (int i = 0; i < 8; ++i) {
    r[i]     = (_Float16)p[i];
    r[i + 8] = (_Float16)p[i + 16];
  }
  return r;
}

__device__ __forceinline__ v8f wmma_f16(v16h a, v16h b, v8f c) {
  return __builtin_amdgcn_wmma_f32_16x16x32_f16(
      false, a, false, b, (short)0, c, false, false);
}

// ---------------------------------------------------------------------------
// Kernel 0: convert the four weight matrices fp32 [K][N] -> f16 transposed
// [N][K] so GEMM B-fragments are contiguous per lane.
// ---------------------------------------------------------------------------
__global__ __launch_bounds__(256) void prep_weights(
    const float* __restrict__ Wq, const float* __restrict__ Wk,
    const float* __restrict__ Wv, const float* __restrict__ Wo,
    _Float16* __restrict__ Wt) {
  int idx = blockIdx.x * blockDim.x + threadIdx.x;  // 4*E*E threads
  int w   = idx / (E * E);
  int rem = idx % (E * E);
  int n = rem / E;
  int k = rem % E;
  const float* W = (w == 0) ? Wq : (w == 1) ? Wk : (w == 2) ? Wv : Wo;
  Wt[(size_t)w * E * E + (size_t)n * E + k] = (_Float16)W[(size_t)k * E + n];
}

// ---------------------------------------------------------------------------
// Kernel 1: fused QKV projections.  grid = (M/256, E/64, 3), block = 256.
// Block tile 256x64; each wave owns 32x64 (8 accumulators, 8 WMMAs/K-step).
// The 64x32 f16 weight tile is async-staged into double-buffered LDS shared
// by all 8 waves; next tile's copy overlaps current tile's WMMAs.
// Q,K written f16 [B,H,T,Dh]; V written transposed f16 [B,H,Dh,T].
// ---------------------------------------------------------------------------
__global__ __launch_bounds__(256) void proj_qkv(
    const float* __restrict__ Xq, const float* __restrict__ Xk,
    const float* __restrict__ Xv, const _Float16* __restrict__ Wt,
    const float* __restrict__ bq, const float* __restrict__ bk,
    const float* __restrict__ bv,
    _Float16* __restrict__ q_out, _Float16* __restrict__ k_out,
    _Float16* __restrict__ vt_out) {
  __shared__ __align__(16) _Float16 wbuf[2][64 * 32];  // [buf][n_local*32+k]

  const int which = blockIdx.z;                    // 0=Q, 1=K, 2=V
  const float* X    = (which == 0) ? Xq : (which == 1) ? Xk : Xv;
  const float* bias = (which == 0) ? bq : (which == 1) ? bk : bv;
  const _Float16* W = Wt + (size_t)which * E * E;

  const int tid  = threadIdx.x;
  const int wave = tid >> 5;
  const int lane = tid & 31;
  const int m0 = blockIdx.x * 256 + wave * 32;
  const int n0 = blockIdx.y * 64;

  // cooperative async fill: 256 threads x 16B = 4KB tile
  const int fnl = tid >> 2;   // 0..63  (local n)
  const int fc  = tid & 3;    // 0..3   (8-half chunk)
  auto fill = [&](int step) {
    async_copy16(W + (size_t)(n0 + fnl) * E + step * 32 + fc * 8,
                 &wbuf[step & 1][fnl * 32 + fc * 8]);
  };

  v8f acc[8] = {};
  auto compute = [&](int step) {
    const int k0 = step * 32;
    __builtin_prefetch(X + (size_t)m0 * E + k0 + 64, 0, 1);
    v16h a0 = frag_ld_f32(X + (size_t)m0 * E + k0, E);
    v16h a1 = frag_ld_f32(X + (size_t)(m0 + 16) * E + k0, E);
    const _Float16* wb = &wbuf[step & 1][0];
    v16h b[4];
#pragma unroll
    for (int j = 0; j < 4; ++j) b[j] = frag_ld_f16(wb + (j * 16) * 32, 32);
#pragma unroll
    for (int j = 0; j < 4; ++j) {
      acc[j]     = wmma_f16(a0, b[j], acc[j]);
      acc[4 + j] = wmma_f16(a1, b[j], acc[4 + j]);
    }
  };

  // software pipeline: branch-free steady state + peeled epilogue
  fill(0);
#pragma unroll 1
  for (int step = 0; step < KSTEPS - 1; ++step) {
    fill(step + 1);
    wait_asynccnt<1>();          // previous fill complete
    __syncthreads();
    compute(step);
    __syncthreads();             // all reads done before buffer is refilled
  }
  wait_asynccnt<0>();
  __syncthreads();
  compute(KSTEPS - 1);

  // C/D layout: component r -> row +r + (lane>=16 ? 8:0), col = lane&15.
  const int mhi = (lane & 16) ? 8 : 0;
  const int nc  = lane & 15;
#pragma unroll
  for (int g = 0; g < 2; ++g) {
#pragma unroll
    for (int j = 0; j < 4; ++j) {
#pragma unroll
      for (int r = 0; r < 8; ++r) {
        int m = m0 + g * 16 + r + mhi;
        int n = n0 + j * 16 + nc;
        _Float16 hv = (_Float16)(acc[g * 4 + j][r] + bias[n]);
        int b = m >> 11;            // m / SEQ
        int t = m & (SEQ - 1);
        int h = n >> 6;             // n / DH
        int d = n & (DH - 1);
        if (which == 2) {
          vt_out[(((size_t)(b * NH + h) * DH + d) * SEQ) + t] = hv;  // [B,H,Dh,T]
        } else {
          _Float16* dst = (which == 0) ? q_out : k_out;
          dst[(((size_t)(b * NH + h) * SEQ + t) * DH) + d] = hv;     // [B,H,T,Dh]
        }
      }
    }
  }
}

// ---------------------------------------------------------------------------
// Kernel 2: flash-style attention.  One wave owns 16 query rows; the 8 waves
// of a block share one head (128 q-tiles per head, 8 per block), so K/V key
// blocks are async-staged into double-buffered LDS once per block:
//   kbuf: 32 keys x 64 d (4KB)   vbuf: 64 d x 32 keys (4KB)
// Per 32-key block: 4 WMMAs for S=QK^T, online softmax (shfl_xor row
// reductions), P relayout via per-wave LDS (s_wait_dscnt), 4 WMMAs for P*V.
// ---------------------------------------------------------------------------
__global__ __launch_bounds__(256) void attn_kernel(
    const _Float16* __restrict__ Q, const _Float16* __restrict__ K,
    const _Float16* __restrict__ Vt, _Float16* __restrict__ Oout) {
  __shared__ __align__(16) _Float16 kbuf[2][32 * 64];
  __shared__ __align__(16) _Float16 vbuf[2][64 * 32];
  __shared__ __align__(16) _Float16 p_lds[8][16 * 32];

  const int tid  = threadIdx.x;
  const int wave = tid >> 5;
  const int lane = tid & 31;
  const int tile = blockIdx.x * 8 + wave;     // 0 .. B*H*(T/16)-1
  const int qt = tile & ((SEQ / 16) - 1);     // query tile within head
  const int bh = tile >> 7;                   // block-uniform (8 | 128)

  const _Float16* Qb = Q  + ((size_t)bh * SEQ + (size_t)qt * 16) * DH;
  const _Float16* Kb = K  + (size_t)bh * SEQ * DH;
  const _Float16* Vb = Vt + (size_t)bh * DH * SEQ;

  // cooperative async fills: each thread moves 16B of K and 16B of V
  const int krow = tid >> 3, kc = tid & 7;    // K: 32 rows x 64 halves
  const int vrow = tid >> 2, vc = tid & 3;    // V: 64 rows x 32 halves
  auto fill = [&](int kb) {
    const int buf = (kb >> 5) & 1;
    async_copy16(Kb + (size_t)(kb + krow) * DH + kc * 8,
                 &kbuf[buf][krow * 64 + kc * 8]);
    async_copy16(Vb + (size_t)vrow * SEQ + kb + vc * 8,
                 &vbuf[buf][vrow * 32 + vc * 8]);
  };

  // Q fragments are invariant over the key sweep: d=0..31 and d=32..63.
  const v16h aq0 = frag_ld_f16(Qb + 0, DH);
  const v16h aq1 = frag_ld_f16(Qb + 32, DH);

  v8f o[4] = {};
  float mrow[8], lrow[8];
#pragma unroll
  for (int r = 0; r < 8; ++r) { mrow[r] = -1e30f; lrow[r] = 0.0f; }

  const float scale = 0.125f;  // 1/sqrt(64)
  const int mhi = (lane & 16) ? 8 : 0;
  const int nc  = lane & 15;
  _Float16* myp = &p_lds[wave][0];

  auto body = [&](int kb) {
    const int buf = (kb >> 5) & 1;
    // ---- scores: two 16x16 tiles covering keys kb..kb+31 (from LDS) ----
    v8f st[2];
#pragma unroll
    for (int t = 0; t < 2; ++t) {
      v16h b0 = frag_ld_f16(&kbuf[buf][(t * 16) * 64 + 0], 64);
      v16h b1 = frag_ld_f16(&kbuf[buf][(t * 16) * 64 + 32], 64);
      v8f s = {};
      s = wmma_f16(aq0, b0, s);
      s = wmma_f16(aq1, b1, s);
      st[t] = s;
    }

    // ---- online softmax; each C-row lives in one VGPR across 16 lanes ----
#pragma unroll
    for (int r = 0; r < 8; ++r) {
      float x0 = st[0][r] * scale;
      float x1 = st[1][r] * scale;
      float mx = fmaxf(x0, x1);
#pragma unroll
      for (int off = 8; off >= 1; off >>= 1)
        mx = fmaxf(mx, __shfl_xor(mx, off, 32));
      float mnew = fmaxf(mrow[r], mx);
      float p0 = __expf(x0 - mnew);
      float p1 = __expf(x1 - mnew);
      float corr = __expf(mrow[r] - mnew);
      float rs = p0 + p1;
#pragma unroll
      for (int off = 8; off >= 1; off >>= 1)
        rs += __shfl_xor(rs, off, 32);
      lrow[r] = lrow[r] * corr + rs;
      mrow[r] = mnew;
#pragma unroll
      for (int j = 0; j < 4; ++j) o[j][r] *= corr;
      int m = r + mhi;                       // scatter P into per-wave tile
      myp[m * 32 + nc]      = (_Float16)p0;
      myp[m * 32 + 16 + nc] = (_Float16)p1;
    }

    wait_dscnt0();               // per-wave DS in order; drain before reload
    v16h ap = frag_ld_f16(myp, 32);          // P as 16x32 A-fragment

#pragma unroll
    for (int j = 0; j < 4; ++j) {            // O += P * V (V^T from LDS)
      v16h bv = frag_ld_f16(&vbuf[buf][(j * 16) * 32], 32);
      o[j] = wmma_f16(ap, bv, o[j]);
    }
  };

  // software pipeline: branch-free steady state + peeled epilogue
  fill(0);
#pragma unroll 1
  for (int kb = 0; kb < SEQ - 32; kb += 32) {
    fill(kb + 32);
    wait_asynccnt<2>();          // previous (K,V) pair complete
    __syncthreads();
    body(kb);
    __syncthreads();             // all waves done reading before refill
  }
  wait_asynccnt<0>();
  __syncthreads();
  body(SEQ - 32);

  // ---- normalize and store f16 row-major [B*T, E] ----
  const int b = bh / NH;
  const int h = bh % NH;
#pragma unroll
  for (int j = 0; j < 4; ++j) {
#pragma unroll
    for (int r = 0; r < 8; ++r) {
      int row = b * SEQ + qt * 16 + r + mhi;
      int col = h * DH + j * 16 + nc;
      Oout[(size_t)row * E + col] = (_Float16)(o[j][r] / lrow[r]);
    }
  }
}

// ---------------------------------------------------------------------------
// Kernel 3: output projection.  attn(f16 [4096,768]) @ Wo^T(f16) + bo -> f32.
// Same async-LDS double-buffered, peeled-pipeline structure as proj_qkv.
// ---------------------------------------------------------------------------
__global__ __launch_bounds__(256) void out_proj(
    const _Float16* __restrict__ A, const _Float16* __restrict__ WtO,
    const float* __restrict__ bo, float* __restrict__ out) {
  __shared__ __align__(16) _Float16 wbuf[2][64 * 32];

  const int tid  = threadIdx.x;
  const int wave = tid >> 5;
  const int lane = tid & 31;
  const int m0 = blockIdx.x * 256 + wave * 32;
  const int n0 = blockIdx.y * 64;

  const int fnl = tid >> 2;
  const int fc  = tid & 3;
  auto fill = [&](int step) {
    async_copy16(WtO + (size_t)(n0 + fnl) * E + step * 32 + fc * 8,
                 &wbuf[step & 1][fnl * 32 + fc * 8]);
  };

  v8f acc[8] = {};
  auto compute = [&](int step) {
    const int k0 = step * 32;
    __builtin_prefetch(A + (size_t)m0 * E + k0 + 32, 0, 1);
    v16h a0 = frag_ld_f16(A + (size_t)m0 * E + k0, E);
    v16h a1 = frag_ld_f16(A + (size_t)(m0 + 16) * E + k0, E);
    const _Float16* wb = &wbuf[step & 1][0];
    v16h b[4];
#pragma unroll
    for (int j = 0; j < 4; ++j) b[j] = frag_ld_f16(wb + (j * 16) * 32, 32);
#pragma unroll
    for (int j = 0; j < 4; ++j) {
      acc[j]     = wmma_f16(a0, b[j], acc[j]);
      acc[4 + j] = wmma_f16(a1, b[j], acc[4 + j]);
    }
  };

  fill(0);
#pragma unroll 1
  for (int step = 0; step < KSTEPS - 1; ++step) {
    fill(step + 1);
    wait_asynccnt<1>();
    __syncthreads();
    compute(step);
    __syncthreads();
  }
  wait_asynccnt<0>();
  __syncthreads();
  compute(KSTEPS - 1);

  const int mhi = (lane & 16) ? 8 : 0;
  const int nc  = lane & 15;
#pragma unroll
  for (int g = 0; g < 2; ++g) {
#pragma unroll
    for (int j = 0; j < 4; ++j) {
#pragma unroll
      for (int r = 0; r < 8; ++r) {
        int m = m0 + g * 16 + r + mhi;
        int n = n0 + j * 16 + nc;
        out[(size_t)m * E + n] = acc[g * 4 + j][r] + bo[n];
      }
    }
  }
}

// ---------------------------------------------------------------------------
// Host-side launch.  Workspace layout (f16 elements):
//   Wt: 4*E*E | Q: B*H*T*Dh | K: B*H*T*Dh | Vt: B*H*Dh*T | attn: (B*T)*E
// Total ~ 29.9 MB.
// ---------------------------------------------------------------------------
extern "C" void kernel_launch(void* const* d_in, const int* in_sizes, int n_in,
                              void* d_out, int out_size, void* d_ws, size_t ws_size,
                              hipStream_t stream) {
  const float* q_in = (const float*)d_in[0];
  const float* k_in = (const float*)d_in[1];
  const float* v_in = (const float*)d_in[2];
  const float* Wq = (const float*)d_in[3];
  const float* bq = (const float*)d_in[4];
  const float* Wk = (const float*)d_in[5];
  const float* bk = (const float*)d_in[6];
  const float* Wv = (const float*)d_in[7];
  const float* bv = (const float*)d_in[8];
  const float* Wo = (const float*)d_in[9];
  const float* bo = (const float*)d_in[10];
  float* out = (float*)d_out;

  _Float16* ws = (_Float16*)d_ws;
  const size_t WT_ELEMS  = 4ull * E * E;
  const size_t QKV_ELEMS = (size_t)BATCH * NH * SEQ * DH;
  _Float16* wt   = ws;
  _Float16* qws  = wt + WT_ELEMS;
  _Float16* kws  = qws + QKV_ELEMS;
  _Float16* vtws = kws + QKV_ELEMS;
  _Float16* attn = vtws + QKV_ELEMS;

  prep_weights<<<dim3((4 * E * E) / 256), dim3(256), 0, stream>>>(Wq, Wk, Wv, Wo, wt);

  proj_qkv<<<dim3(M_TOTAL / 256, E / 64, 3), dim3(256), 0, stream>>>(
      q_in, k_in, v_in, wt, bq, bk, bv, qws, kws, vtws);

  attn_kernel<<<dim3((BATCH * NH * (SEQ / 16)) / 8), dim3(256), 0, stream>>>(
      qws, kws, vtws, attn);

  out_proj<<<dim3(M_TOTAL / 256, E / 64), dim3(256), 0, stream>>>(
      attn, wt + 3ull * E * E, bo, out);
}
